// EquivariantAttention_78718160601521
// MI455X (gfx1250) — compile-verified, matrix operand order checked by reading, not run on hardware
//
#include <hip/hip_runtime.h>

// ---------------------------------------------------------------------------
// EquivariantAttention for MI455X (gfx1250): bf16 WMMA for all GEMM-shaped
// work (QKV proj, DPB MLP, output proj), fp32 VALU for LN/softmax/coord math.
// ---------------------------------------------------------------------------

typedef __attribute__((ext_vector_type(16))) __bf16 v16bf;
typedef __attribute__((ext_vector_type(16))) unsigned short v16us;
typedef __attribute__((ext_vector_type(8))) float v8f;

constexpr int B_ = 4, N_ = 2048, DIM_ = 256, H_ = 4, DH_ = 64, INNER_ = 256;
constexpr int KNN = 32, PD = 128;   // NUM_NN, PDIM

__device__ __forceinline__ unsigned short f2bf(float f) {
  union { float f; unsigned u; } x; x.f = f;
  unsigned r = x.u + 0x7FFFu + ((x.u >> 16) & 1u);   // round-to-nearest-even
  return (unsigned short)(r >> 16);
}

// Load one 16x32 bf16 WMMA operand fragment (A from row-major activations, or
// B from a pre-transposed weight W^T so the 8-element runs are contiguous).
// src points at (row0, k0); ld = row stride in elements.
// Layout (ISA 7.12.2, 16-bit A 16x32): lanes 0-15 row=lane K=0..7,16..23;
// lanes 16-31 row=lane-16 K=8..15,24..31.
__device__ __forceinline__ v16us load_frag(const unsigned short* src, int ld, int lane) {
  const unsigned short* p = src + (lane & 15) * ld + ((lane >> 4) << 3);
  v16us a;
#pragma unroll
  for (int e = 0; e < 8; ++e) a[e] = p[e];
#pragma unroll
  for (int e = 0; e < 8; ++e) a[8 + e] = p[16 + e];
  return a;
}

__device__ __forceinline__ v8f wmma_bf16(v16us a, v16us b, v8f c) {
  return __builtin_amdgcn_wmma_f32_16x16x32_bf16(
      false, __builtin_bit_cast(v16bf, a),
      false, __builtin_bit_cast(v16bf, b),
      (short)0, c, false, false);
}

// ---------------------------------------------------------------------------
// Kernel 1: LayerNorm(feats) * ln_g  -> bf16  (one row of 256 per block)
// ---------------------------------------------------------------------------
__global__ void __launch_bounds__(256)
k_ln(const float* __restrict__ feats, const float* __restrict__ g,
     unsigned short* __restrict__ outbf) {
  __shared__ float sm[DIM_];
  const int t = threadIdx.x;
  const size_t row = blockIdx.x;
  float v = feats[row * DIM_ + t];
  sm[t] = v;
  __syncthreads();
  for (int s = DIM_ / 2; s > 0; s >>= 1) { if (t < s) sm[t] += sm[t + s]; __syncthreads(); }
  float mean = sm[0] * (1.f / DIM_);
  __syncthreads();
  float d = v - mean;
  sm[t] = d * d;
  __syncthreads();
  for (int s = DIM_ / 2; s > 0; s >>= 1) { if (t < s) sm[t] += sm[t + s]; __syncthreads(); }
  float var = sm[0] * (1.f / DIM_);
  outbf[row * DIM_ + t] = f2bf(d * rsqrtf(var + 1e-5f) * g[t]);
}

// ---------------------------------------------------------------------------
// Kernel 2: fp32 [K,Nn] -> bf16 transposed [Nn,K]
// ---------------------------------------------------------------------------
__global__ void k_f2bf_t(const float* __restrict__ src, unsigned short* __restrict__ dst,
                         int K, int Nn) {
  int t = blockIdx.x * blockDim.x + threadIdx.x;
  if (t < K * Nn) {
    int k = t / Nn, c = t - k * Nn;
    dst[(size_t)c * K + k] = f2bf(src[t]);
  }
}

// dpb_qk_w [128,4] -> bf16 transposed + zero-padded to [16,128]
__global__ void k_padqk_t(const float* __restrict__ w, unsigned short* __restrict__ dst) {
  int t = blockIdx.x * blockDim.x + threadIdx.x;   // t = c*128 + k
  if (t < 16 * PD) {
    int c = t >> 7;
    int k = t & (PD - 1);
    dst[t] = (c < H_) ? f2bf(w[k * H_ + c]) : (unsigned short)0;
  }
}

// ---------------------------------------------------------------------------
// Kernel 3/6: generic bf16 GEMM  D[M,Nn] = A[M,K] @ B[K,Nn] (+bias)
// A row-major bf16, BT = B^T row-major bf16 [Nn,K]. 4 waves/block, each wave
// computes a 16x64 tile (A fragment reused across 4 WMMAs per k-step).
// ---------------------------------------------------------------------------
__global__ void __launch_bounds__(128)
k_gemm_bf16(const unsigned short* __restrict__ A,
            const unsigned short* __restrict__ BT,
            const float* __restrict__ bias,
            float* __restrict__ D, int M, int Nn, int K) {
  const int lane = threadIdx.x & 31;
  const int wave = threadIdx.x >> 5;
  const int row0 = (blockIdx.y * 4 + wave) * 16;
  const int col0 = blockIdx.x * 64;
  if (row0 >= M) return;
  v8f acc[4] = {};
  for (int k0 = 0; k0 < K; k0 += 32) {
    v16us a = load_frag(A + (size_t)row0 * K + k0, K, lane);
#pragma unroll
    for (int t = 0; t < 4; ++t) {
      v16us b = load_frag(BT + (size_t)(col0 + t * 16) * K + k0, K, lane);
      acc[t] = wmma_bf16(a, b, acc[t]);
    }
  }
  const int col = lane & 15, half = lane >> 4;
#pragma unroll
  for (int t = 0; t < 4; ++t) {
    int c0 = col0 + t * 16 + col;
    float bv = bias ? bias[c0] : 0.f;
    float* dst = D + (size_t)row0 * Nn + c0;
#pragma unroll
    for (int r = 0; r < 8; ++r) dst[(size_t)(r + half * 8) * Nn] = acc[t][r] + bv;
  }
}

// ---------------------------------------------------------------------------
// Kernel 4: 32 nearest neighbors per point. Coords for the batch live in LDS
// (24KB); each thread keeps a sorted 32-candidate list in LDS. Strict '<'
// insertion reproduces top_k's ascending-index tie-break.
// ---------------------------------------------------------------------------
__global__ void __launch_bounds__(128)
k_topk(const float* __restrict__ coors, int* __restrict__ idx,
       float* __restrict__ dist, float* __restrict__ relc) {
  __shared__ float sc[N_ * 3];
  __shared__ float sld[128][KNN];
  __shared__ int   sli[128][KNN];
  const int b = blockIdx.y;
  const int i = blockIdx.x * 128 + threadIdx.x;
  for (int t = threadIdx.x; t < N_ * 3; t += 128)
    sc[t] = coors[(size_t)b * N_ * 3 + t];
  float* myd = sld[threadIdx.x];
  int*   myi = sli[threadIdx.x];
#pragma unroll
  for (int t = 0; t < KNN; ++t) { myd[t] = 3.4e38f; myi[t] = 0; }
  __syncthreads();
  const float xi = sc[i * 3 + 0], yi = sc[i * 3 + 1], zi = sc[i * 3 + 2];
  for (int j = 0; j < N_; ++j) {
    float dx = xi - sc[j * 3 + 0];
    float dy = yi - sc[j * 3 + 1];
    float dz = zi - sc[j * 3 + 2];
    float d2 = dx * dx + dy * dy + dz * dz;
    if (d2 < myd[KNN - 1]) {
      int p = KNN - 1;
      while (p > 0 && d2 < myd[p - 1]) {
        myd[p] = myd[p - 1]; myi[p] = myi[p - 1]; --p;
      }
      myd[p] = d2; myi[p] = j;
    }
  }
  const size_t base = ((size_t)b * N_ + i) * KNN;
  for (int t = 0; t < KNN; ++t) {
    int j = myi[t];
    idx[base + t]  = j;
    dist[base + t] = sqrtf(myd[t]);
    relc[(base + t) * 3 + 0] = xi - sc[j * 3 + 0];
    relc[(base + t) * 3 + 1] = yi - sc[j * 3 + 1];
    relc[(base + t) * 3 + 2] = zi - sc[j * 3 + 2];
  }
}

// ---------------------------------------------------------------------------
// Kernel 5 helpers: per-wave 16-row MLP tile (WMMA) + LayerNorm/SiLU staging.
// ---------------------------------------------------------------------------
__device__ __forceinline__ void ln_silu(float* xsw, unsigned short* xbw,
                                        const float* g, const float* be,
                                        float* mu, float* rs, int lane) {
  __syncthreads();
  if (lane < 16) {
    const float* row = xsw + lane * PD;
    float s = 0.f, ss = 0.f;
    for (int c = 0; c < PD; ++c) { float v = row[c]; s += v; ss += v * v; }
    float m = s * (1.f / PD);
    mu[lane] = m;
    rs[lane] = rsqrtf(fmaxf(ss * (1.f / PD) - m * m, 0.f) + 1e-5f);
  }
  __syncthreads();
  for (int t = lane; t < 16 * PD; t += 32) {
    int r = t >> 7, c = t & (PD - 1);
    float v = (xsw[t] - mu[r]) * rs[r] * g[c] + be[c];
    float sl = v / (1.f + __expf(-v));   // SiLU
    xbw[t] = f2bf(sl);
  }
  __syncthreads();
}

// xs[16,128] = bf16(xb)[16,128] @ W[128,128] + bias   (WT = W^T bf16)
__device__ __forceinline__ void mlp128(const unsigned short* xbw, float* xsw,
                                       const unsigned short* WT, const float* bias,
                                       int lane) {
  v16us a0 = load_frag(xbw + 0,  PD, lane);
  v16us a1 = load_frag(xbw + 32, PD, lane);
  v16us a2 = load_frag(xbw + 64, PD, lane);
  v16us a3 = load_frag(xbw + 96, PD, lane);
  const int col = lane & 15, half = lane >> 4;
  for (int ct = 0; ct < 8; ++ct) {
    v8f acc = {};
    acc = wmma_bf16(a0, load_frag(WT + (size_t)(ct * 16) * PD + 0,  PD, lane), acc);
    acc = wmma_bf16(a1, load_frag(WT + (size_t)(ct * 16) * PD + 32, PD, lane), acc);
    acc = wmma_bf16(a2, load_frag(WT + (size_t)(ct * 16) * PD + 64, PD, lane), acc);
    acc = wmma_bf16(a3, load_frag(WT + (size_t)(ct * 16) * PD + 96, PD, lane), acc);
    int c0 = ct * 16 + col;
    float bv = bias[c0];
#pragma unroll
    for (int r = 0; r < 8; ++r) xsw[(r + half * 8) * PD + c0] = acc[r] + bv;
  }
}

// ---------------------------------------------------------------------------
// Kernel 5: fused per-(b,i) DPB-MLP (WMMA) + attention + coordinate branch.
// Block = 64 threads (2 waves); wave w owns neighbor rows w*16..w*16+15.
// ---------------------------------------------------------------------------
__global__ void __launch_bounds__(64)
k_attn(const float* __restrict__ qkv,          // [B*N, 768] (q|k|v)
       const int*   __restrict__ nidx,         // [B*N, 32]
       const float* __restrict__ ndist,        // [B*N, 32]
       const float* __restrict__ nrelc,        // [B*N, 32, 3]
       const float* __restrict__ w1, const float* __restrict__ b1,
       const float* __restrict__ g1, const float* __restrict__ be1,
       const unsigned short* __restrict__ w2T, const float* __restrict__ b2,
       const float* __restrict__ g2, const float* __restrict__ be2,
       const unsigned short* __restrict__ w3T, const float* __restrict__ b3,
       const float* __restrict__ g3, const float* __restrict__ be3,
       const unsigned short* __restrict__ qkT, const float* __restrict__ qkb,
       const unsigned short* __restrict__ vwT, const float* __restrict__ vb,
       const float* __restrict__ cmw1, const float* __restrict__ cmw2,
       const float* __restrict__ gw,   const float* __restrict__ gb,
       const float* __restrict__ cns,  const float* __restrict__ comb,
       unsigned short* __restrict__ node_bf,   // [B*N, 256]
       float* __restrict__ out_coors) {        // [B*N, 3]
  __shared__ float          xs[2][16][PD];
  __shared__ unsigned short xb[2][16][PD];
  __shared__ float qs[INNER_], qn[INNER_], rqn[H_];
  __shared__ float posv[2][16], mu16[2][16], rs16[2][16];
  __shared__ float qkl[H_][KNN], attnw[H_][KNN];
  __shared__ float vp[KNN][INNER_];
  __shared__ float cw[KNN][H_], sgn[KNN][H_], rcs[KNN][3];
  __shared__ int   idxs[KNN];

  const int tid  = threadIdx.x;
  const int lane = tid & 31;
  const int wave = tid >> 5;
  const long bi    = blockIdx.x;           // b*N + i
  const long bbase = (bi >> 11) << 11;     // b*N

  for (int t = tid; t < INNER_; t += 64) qs[t] = qkv[bi * 768 + t];
  if (tid < KNN) idxs[tid] = nidx[bi * KNN + tid];
  if (lane < 16) posv[wave][lane] = -100.f * ndist[bi * KNN + wave * 16 + lane];
  __syncthreads();
  if (tid < H_) {
    float s = 0.f;
    for (int d = 0; d < DH_; ++d) { float v = qs[tid * DH_ + d]; s += v * v; }
    rqn[tid] = 1.f / fmaxf(sqrtf(s), 1e-12f);
  }
  __syncthreads();
  for (int t = tid; t < INNER_; t += 64) qn[t] = qs[t] * rqn[t >> 6];

  float*          xsw = &xs[wave][0][0];
  unsigned short* xbw = &xb[wave][0][0];

  // DPB layer 1: scalar pos broadcast into 128 hidden units
  for (int t = lane; t < 16 * PD; t += 32) {
    int r = t >> 7, c = t & (PD - 1);
    xsw[t] = posv[wave][r] * w1[c] + b1[c];
  }
  ln_silu(xsw, xbw, g1, be1, mu16[wave], rs16[wave], lane);
  mlp128(xbw, xsw, w2T, b2, lane);
  ln_silu(xsw, xbw, g2, be2, mu16[wave], rs16[wave], lane);
  mlp128(xbw, xsw, w3T, b3, lane);
  ln_silu(xsw, xbw, g3, be3, mu16[wave], rs16[wave], lane);

  // Projections from the final activation (bf16 in xbw)
  v16us a0 = load_frag(xbw + 0,  PD, lane);
  v16us a1 = load_frag(xbw + 32, PD, lane);
  v16us a2 = load_frag(xbw + 64, PD, lane);
  v16us a3 = load_frag(xbw + 96, PD, lane);
  const int col = lane & 15, half = lane >> 4;

  { // qk_pos: [16,128] @ [128,4] (N padded to 16)
    v8f acc = {};
    acc = wmma_bf16(a0, load_frag(qkT + 0,  PD, lane), acc);
    acc = wmma_bf16(a1, load_frag(qkT + 32, PD, lane), acc);
    acc = wmma_bf16(a2, load_frag(qkT + 64, PD, lane), acc);
    acc = wmma_bf16(a3, load_frag(qkT + 96, PD, lane), acc);
    if (col < H_) {
      float bv = qkb[col];
#pragma unroll
      for (int r = 0; r < 8; ++r)
        qkl[col][wave * 16 + r + half * 8] = acc[r] + bv;
    }
  }
  for (int ct = 0; ct < 16; ++ct) {  // value_pos: [16,128] @ [128,256]
    v8f acc = {};
    acc = wmma_bf16(a0, load_frag(vwT + (size_t)(ct * 16) * PD + 0,  PD, lane), acc);
    acc = wmma_bf16(a1, load_frag(vwT + (size_t)(ct * 16) * PD + 32, PD, lane), acc);
    acc = wmma_bf16(a2, load_frag(vwT + (size_t)(ct * 16) * PD + 64, PD, lane), acc);
    acc = wmma_bf16(a3, load_frag(vwT + (size_t)(ct * 16) * PD + 96, PD, lane), acc);
    int c0 = ct * 16 + col;
    float bv = vb[c0];
#pragma unroll
    for (int r = 0; r < 8; ++r)
      vp[wave * 16 + r + half * 8][c0] = acc[r] + bv;
  }
  __syncthreads();

  // logits: qk = 8 * <q_hat, k_hat> + qk_pos      (128 (h,nn) pairs / 64 thr)
  for (int p = tid; p < H_ * KNN; p += 64) {
    int h = p >> 5, nn = p & (KNN - 1);
    const float* kp = qkv + (bbase + idxs[nn]) * 768 + 256 + h * DH_;
    const float* qp = qn + h * DH_;
    float ss = 0.f, dt = 0.f;
    for (int d = 0; d < DH_; ++d) { float kv = kp[d]; ss += kv * kv; dt += qp[d] * kv; }
    float rk = 1.f / fmaxf(sqrtf(ss), 1e-12f);
    qkl[h][nn] = dt * rk * 8.0f + qkl[h][nn];
  }
  __syncthreads();

  // per-j coordinate MLP (+ node-attn softmax in spare threads)
  if (tid < KNN) {
    int j = tid;
    float c0 = qkl[0][j], c1 = qkl[1][j], c2 = qkl[2][j], c3 = qkl[3][j];
    float hid[16];
#pragma unroll
    for (int c = 0; c < 16; ++c) {
      float hsum = c0 * cmw1[c] + c1 * cmw1[16 + c] + c2 * cmw1[32 + c] + c3 * cmw1[48 + c];
      hid[c] = 0.5f * hsum * (1.f + erff(hsum * 0.70710678f));   // exact gelu
    }
#pragma unroll
    for (int hh = 0; hh < H_; ++hh) {
      float s = 0.f;
#pragma unroll
      for (int c = 0; c < 16; ++c) s += hid[c] * cmw2[c * H_ + hh];
      cw[j][hh] = s;
      float tg = c0 * gw[hh] + c1 * gw[H_ + hh] + c2 * gw[2 * H_ + hh] + c3 * gw[3 * H_ + hh] + gb[hh];
      sgn[j][hh] = tanhf(tg);
    }
    float dx = nrelc[(bi * KNN + j) * 3 + 0];
    float dy = nrelc[(bi * KNN + j) * 3 + 1];
    float dz = nrelc[(bi * KNN + j) * 3 + 2];
    float rn = sqrtf(dx * dx + dy * dy + dz * dz);
    float sc = cns[0] / fmaxf(rn, 1e-8f);
    rcs[j][0] = dx * sc; rcs[j][1] = dy * sc; rcs[j][2] = dz * sc;
  } else if (tid < KNN + H_) {
    int h = tid - KNN;                       // node softmax over nn
    float mx = -3.4e38f;
    for (int j = 0; j < KNN; ++j) mx = fmaxf(mx, qkl[h][j]);
    float s = 0.f;
    for (int j = 0; j < KNN; ++j) { float e = __expf(qkl[h][j] - mx); attnw[h][j] = e; s += e; }
    float r = 1.f / s;
    for (int j = 0; j < KNN; ++j) attnw[h][j] *= r;
  }
  __syncthreads();
  if (tid < H_) {                            // coordinate softmax over j
    int h = tid;
    float mx = -3.4e38f;
    for (int j = 0; j < KNN; ++j) mx = fmaxf(mx, cw[j][h]);
    float s = 0.f;
    for (int j = 0; j < KNN; ++j) { float e = __expf(cw[j][h] - mx); cw[j][h] = e; s += e; }
    float r = 1.f / s;
    for (int j = 0; j < KNN; ++j) cw[j][h] *= r;
  }
  __syncthreads();

  if (tid < 3) {                             // coors_out[b,i,c]
    int c = tid;
    float o = 0.f;
    for (int h = 0; h < H_; ++h) {
      float s = 0.f;
      for (int j = 0; j < KNN; ++j) s += cw[j][h] * rcs[j][c] * sgn[j][h];
      o += s * comb[h];
    }
    out_coors[bi * 3 + c] = o;
  }
  for (int p = tid; p < INNER_; p += 64) {   // node out = attn @ (v + value_pos)
    int h = p >> 6;
    float o = 0.f;
    for (int nn = 0; nn < KNN; ++nn) {
      long j = bbase + idxs[nn];
      o += attnw[h][nn] * (qkv[j * 768 + 512 + p] + vp[nn][p]);
    }
    node_bf[bi * INNER_ + p] = f2bf(o);
  }
}

// ---------------------------------------------------------------------------
extern "C" void kernel_launch(void* const* d_in, const int* in_sizes, int n_in,
                              void* d_out, int out_size, void* d_ws, size_t ws_size,
                              hipStream_t stream) {
  const float* feats    = (const float*)d_in[0];
  const float* coors    = (const float*)d_in[1];
  const float* ln_g     = (const float*)d_in[2];
  const float* w_qkv    = (const float*)d_in[3];
  const float* w_out    = (const float*)d_in[4];
  const float* b_out    = (const float*)d_in[5];
  const float* dpb_w1   = (const float*)d_in[6];
  const float* dpb_b1   = (const float*)d_in[7];
  const float* dpb_g1   = (const float*)d_in[8];
  const float* dpb_be1  = (const float*)d_in[9];
  const float* dpb_w2   = (const float*)d_in[10];
  const float* dpb_b2   = (const float*)d_in[11];
  const float* dpb_g2   = (const float*)d_in[12];
  const float* dpb_be2  = (const float*)d_in[13];
  const float* dpb_w3   = (const float*)d_in[14];
  const float* dpb_b3   = (const float*)d_in[15];
  const float* dpb_g3   = (const float*)d_in[16];
  const float* dpb_be3  = (const float*)d_in[17];
  const float* dpb_qk_w = (const float*)d_in[18];
  const float* dpb_qk_b = (const float*)d_in[19];
  const float* dpb_v_w  = (const float*)d_in[20];
  const float* dpb_v_b  = (const float*)d_in[21];
  const float* cm_w1    = (const float*)d_in[22];
  const float* cm_w2    = (const float*)d_in[23];
  const float* gate_w   = (const float*)d_in[24];
  const float* gate_b   = (const float*)d_in[25];
  const float* cn_scale = (const float*)d_in[26];
  const float* combine  = (const float*)d_in[27];

  float* out       = (float*)d_out;                       // [B,N,256]
  float* out_coors = out + (size_t)B_ * N_ * DIM_;        // [B,N,3]

  // workspace carve-up (~40 MB total)
  char*  ws  = (char*)d_ws;
  size_t off = 0;
  auto alloc = [&](size_t bytes) {
    char* p = ws + off;
    off += (bytes + 255) & ~(size_t)255;
    return p;
  };
  const size_t MN = (size_t)B_ * N_;                      // 8192 rows
  float*          qkv      = (float*)         alloc(MN * 768 * 4);
  unsigned short* feats_bf = (unsigned short*)alloc(MN * DIM_ * 2);
  unsigned short* wqkvT    = (unsigned short*)alloc((size_t)768 * DIM_ * 2);
  unsigned short* woutT    = (unsigned short*)alloc((size_t)DIM_ * INNER_ * 2);
  unsigned short* w2T      = (unsigned short*)alloc((size_t)PD * PD * 2);
  unsigned short* w3T      = (unsigned short*)alloc((size_t)PD * PD * 2);
  unsigned short* vwT      = (unsigned short*)alloc((size_t)INNER_ * PD * 2);
  unsigned short* qkT      = (unsigned short*)alloc((size_t)16 * PD * 2);
  int*            nidx     = (int*)           alloc(MN * KNN * 4);
  float*          ndist    = (float*)         alloc(MN * KNN * 4);
  float*          nrelc    = (float*)         alloc(MN * KNN * 3 * 4);
  unsigned short* node_bf  = (unsigned short*)alloc(MN * INNER_ * 2);
  (void)ws_size; (void)in_sizes; (void)n_in; (void)out_size;

  // 1) LayerNorm feats -> bf16
  k_ln<<<(int)MN, 256, 0, stream>>>(feats, ln_g, feats_bf);

  // 2) bf16 transposed weight copies
  auto cvt = [&](const float* s, unsigned short* d, int K, int Nn) {
    int n = K * Nn;
    k_f2bf_t<<<(n + 255) / 256, 256, 0, stream>>>(s, d, K, Nn);
  };
  cvt(w_qkv,   wqkvT, DIM_, 768);
  cvt(w_out,   woutT, INNER_, DIM_);
  cvt(dpb_w2,  w2T,   PD, PD);
  cvt(dpb_w3,  w3T,   PD, PD);
  cvt(dpb_v_w, vwT,   PD, INNER_);
  k_padqk_t<<<(16 * PD + 255) / 256, 256, 0, stream>>>(dpb_qk_w, qkT);

  // 3) QKV projection: [8192,256] @ [256,768] (WMMA)
  k_gemm_bf16<<<dim3(768 / 64, (unsigned)(MN / 64)), 128, 0, stream>>>(
      feats_bf, wqkvT, nullptr, qkv, (int)MN, 768, DIM_);

  // 4) 32-NN search
  k_topk<<<dim3(N_ / 128, B_), 128, 0, stream>>>(coors, nidx, ndist, nrelc);

  // 5) fused DPB MLP (WMMA) + attention + coordinate branch
  k_attn<<<(int)MN, 64, 0, stream>>>(
      qkv, nidx, ndist, nrelc,
      dpb_w1, dpb_b1, dpb_g1, dpb_be1,
      w2T, dpb_b2, dpb_g2, dpb_be2,
      w3T, dpb_b3, dpb_g3, dpb_be3,
      qkT, dpb_qk_b, vwT, dpb_v_b,
      cm_w1, cm_w2, gate_w, gate_b, cn_scale, combine,
      node_bf, out_coors);

  // 6) output projection: [8192,256] @ [256,256] + b_out (WMMA)
  k_gemm_bf16<<<dim3(DIM_ / 64, (unsigned)(MN / 64)), 128, 0, stream>>>(
      node_bf, woutT, b_out, out, (int)MN, DIM_, INNER_);
}
